// NamedEntityRecognitionModel_80985903334131
// MI455X (gfx1250) — matrix operand order repaired and checked by reading, not compiled, not run on hardware
//
#include <hip/hip_runtime.h>

// ---------------------------------------------------------------------------
// BiLSTM-CRF NER loss for MI455X (gfx1250, wave32, WMMA).
// GEMMs use v_wmma_f32_16x16x32_f16 with pre-packed B fragments.
// LSTM step stages weight fragments into LDS via async-to-LDS loads
// (ASYNCcnt path) with 4x reuse across batch tiles.
// ---------------------------------------------------------------------------

typedef __attribute__((ext_vector_type(16))) _Float16 v16h;
typedef __attribute__((ext_vector_type(8)))  float    v8f;

#define Bn   64
#define Tn   512
#define En   256
#define Hn   256
#define HIDn 512
#define Kt   32   // tags

struct F16x16 { uint4 lo, hi; };

__device__ __forceinline__ v16h load_frag_a(const _Float16* p0, const _Float16* p1) {
  F16x16 t;
  t.lo = *(const uint4*)p0;   // elements j=0..7   (16B aligned)
  t.hi = *(const uint4*)p1;   // elements j=8..15  (16B aligned)
  return __builtin_bit_cast(v16h, t);
}

// ---------------------------------------------------------------------------
// Pack LSTM weights: B[k][n] = (k<256 ? Wi[n][k] : Wh[n][k-256]), K=512, N=1024.
// Fragment layout (f16 B 32x16): lane holds col n=lane&15, K = 8*(lane>>4)+j (j<8), 16+8*(lane>>4)+(j-8) (j>=8).
// tile linear index = ntile*16 + kc ; per-tile 32 lanes * 16 contiguous halves.
// ---------------------------------------------------------------------------
__global__ void pack_lstm_weights(const float* __restrict__ Wi,
                                  const float* __restrict__ Wh,
                                  _Float16* __restrict__ dst) {
  int tid  = blockIdx.x * blockDim.x + threadIdx.x;
  int lane = tid & 31;
  int tile = tid >> 5;                 // 0..1023  (64 ntiles * 16 kchunks)
  if (tile >= 64 * 16) return;
  int kc    = tile & 15;
  int ntile = tile >> 4;
  int n     = ntile * 16 + (lane & 15);
  int kbase = kc * 32 + (lane >> 4) * 8;
  _Float16* out = dst + (size_t)tile * 512 + lane * 16;
#pragma unroll
  for (int j = 0; j < 16; ++j) {
    int k = kbase + (j < 8 ? j : j + 8);
    float v = (k < En) ? Wi[(size_t)n * En + k] : Wh[(size_t)n * Hn + (k - En)];
    out[j] = (_Float16)v;
  }
}

// fcW [32, 512] -> packed B fragments, K=512 (16 kchunks), N=32 (2 ntiles)
__global__ void pack_fc_weights(const float* __restrict__ fcW,
                                _Float16* __restrict__ dst) {
  int tid  = blockIdx.x * blockDim.x + threadIdx.x;
  int lane = tid & 31;
  int tile = tid >> 5;                 // 0..31 (2 ntiles * 16 kchunks)
  if (tile >= 2 * 16) return;
  int kc    = tile & 15;
  int ntile = tile >> 4;
  int n     = ntile * 16 + (lane & 15);
  int kbase = kc * 32 + (lane >> 4) * 8;
  _Float16* out = dst + (size_t)tile * 512 + lane * 16;
#pragma unroll
  for (int j = 0; j < 16; ++j)
    out[j] = (_Float16)fcW[(size_t)n * HIDn + kbase + (j < 8 ? j : j + 8)];
}

// Embedding gather + fp32 -> f16
__global__ void embed_gather(const int* __restrict__ X,
                             const float* __restrict__ embed,
                             _Float16* __restrict__ xf) {
  long long i = (long long)blockIdx.x * 256 + threadIdx.x;  // B*T*E elements
  if (i >= (long long)Bn * Tn * En) return;
  int e = (int)(i & (En - 1));
  long long bt = i >> 8;
  int tok = X[bt];
  xf[i] = (_Float16)embed[(size_t)tok * En + e];
}

// ---------------------------------------------------------------------------
// One LSTM timestep, both directions (blockIdx.y = dir).
// g = [x_t, h_{t-1}] @ Wcat^T + b ; gates fused in-register via 4 accumulators.
// Grid: dim3(16, 2), block 128 (4 waves = batch tiles mt 0..3), blockIdx.x = jt.
// Weight tiles [gate][kc] (64 KB) staged block-wide into LDS with async loads,
// each tile reused by the 4 batch-tile waves.
// ---------------------------------------------------------------------------
__global__ __launch_bounds__(128)
void lstm_step(int t,
               const _Float16* __restrict__ xf,      // [B][T][E] f16
               const _Float16* __restrict__ wpack,   // [2][1024 tiles][512]
               const float* __restrict__ bias_f,
               const float* __restrict__ bias_b,
               const _Float16* __restrict__ h_in,    // [2][B][H] f16
               _Float16* __restrict__ h_out,         // [2][B][H] f16
               float* __restrict__ c_state,          // [2][B][H] f32
               _Float16* __restrict__ H_all) {       // [B][T][HID] f16
  __shared__ _Float16 wsm[4 * 16 * 512];              // 64 KB: [gate][kc] tiles
  int dir  = blockIdx.y;
  int lane = threadIdx.x & 31;
  int wv   = threadIdx.x >> 5;                      // 0..3 = batch tile mt
  int mt   = wv;
  int jt   = blockIdx.x;                            // 0..15 hidden-j tile
  int tt   = dir ? (Tn - 1 - t) : t;

  const _Float16* wp   = wpack + (size_t)dir * (1024 * 512);
  const float*    bias = dir ? bias_b : bias_f;
  const _Float16* hs   = h_in + (size_t)dir * Bn * Hn;

  int rowA  = mt * 16 + (lane & 15);                // batch row this lane loads
  int base8 = (lane >> 4) * 8;                      // A-frag K sub-offset

  // Prefetch next timestep's embedding row for this lane (global_prefetch_b8).
  {
    int tnext = dir ? (tt - 1) : (tt + 1);
    if (tnext >= 0 && tnext < Tn)
      __builtin_prefetch(xf + ((size_t)rowA * Tn + tnext) * En + base8, 0, 1);
  }

  // ---- Stage 64 weight tiles (1 KB each) into LDS via async-to-LDS loads.
  // Wave wv stages tiles idx = wv*16 .. wv*16+15; per tile each lane moves
  // 32 contiguous bytes (two b128 transfers).
#pragma unroll 4
  for (int i = 0; i < 16; ++i) {
    int idx = wv * 16 + i;                          // 0..63
    int sg  = idx >> 4;                             // gate
    int skc = idx & 15;                             // k-chunk
    const _Float16* src =
        wp + (((size_t)(sg * 16 + jt) * 16 + skc) * 32 + lane) * 16;
    unsigned lds = (unsigned)(uintptr_t)(&wsm[((size_t)idx * 32 + lane) * 16]);
    unsigned long long ga = (unsigned long long)(uintptr_t)src;
    asm volatile("global_load_async_to_lds_b128 %0, %1, off"
                 :: "v"(lds), "v"(ga) : "memory");
    asm volatile("global_load_async_to_lds_b128 %0, %1, off"
                 :: "v"(lds + 16u), "v"(ga + 16ull) : "memory");
  }
  asm volatile("s_wait_asynccnt 0x0" ::: "memory");
  __syncthreads();

  v8f acc0 = {}, acc1 = {}, acc2 = {}, acc3 = {};

  for (int kc = 0; kc < 16; ++kc) {
    const _Float16* ap;
    if (kc < 8)
      ap = xf + ((size_t)rowA * Tn + tt) * En + kc * 32 + base8;
    else
      ap = hs + (size_t)rowA * Hn + (kc - 8) * 32 + base8;
    v16h a = load_frag_a(ap, ap + 16);

    v16h b0 = *(const v16h*)(wsm + ((size_t)(0 * 16 + kc) * 512 + lane * 16));
    v16h b1 = *(const v16h*)(wsm + ((size_t)(1 * 16 + kc) * 512 + lane * 16));
    v16h b2 = *(const v16h*)(wsm + ((size_t)(2 * 16 + kc) * 512 + lane * 16));
    v16h b3 = *(const v16h*)(wsm + ((size_t)(3 * 16 + kc) * 512 + lane * 16));

    acc0 = __builtin_amdgcn_wmma_f32_16x16x32_f16(false, a, false, b0, (short)0, acc0, false, false);
    acc1 = __builtin_amdgcn_wmma_f32_16x16x32_f16(false, a, false, b1, (short)0, acc1, false, false);
    acc2 = __builtin_amdgcn_wmma_f32_16x16x32_f16(false, a, false, b2, (short)0, acc2, false, false);
    acc3 = __builtin_amdgcn_wmma_f32_16x16x32_f16(false, a, false, b3, (short)0, acc3, false, false);
  }

  int colj = jt * 16 + (lane & 15);
  float bi = bias[colj];
  float bf = bias[Hn + colj];
  float bg = bias[2 * Hn + colj];
  float bo = bias[3 * Hn + colj];

#pragma unroll
  for (int r = 0; r < 8; ++r) {
    int row = mt * 16 + r + ((lane >> 4) << 3);     // batch index
    float iv = 1.0f / (1.0f + __expf(-(acc0[r] + bi)));
    float fv = 1.0f / (1.0f + __expf(-(acc1[r] + bf)));
    float gv = tanhf(acc2[r] + bg);
    float ov = 1.0f / (1.0f + __expf(-(acc3[r] + bo)));
    size_t ci = (size_t)dir * Bn * Hn + (size_t)row * Hn + colj;
    float c = fv * c_state[ci] + iv * gv;
    c_state[ci] = c;
    float h = ov * tanhf(c);
    h_out[ci] = (_Float16)h;
    H_all[((size_t)row * Tn + tt) * HIDn + (size_t)dir * Hn + colj] = (_Float16)h;
  }
}

// Emissions: em = H_all @ fcW^T + fcb.  M=32768, N=32, K=512.
__global__ __launch_bounds__(256)
void fc_emissions(const _Float16* __restrict__ H_all,
                  const _Float16* __restrict__ fcp,
                  const float* __restrict__ fcb,
                  float* __restrict__ em) {
  int lane = threadIdx.x & 31;
  int wid  = blockIdx.x * 8 + (threadIdx.x >> 5);   // 0..4095
  int mt   = wid >> 1;
  int nt   = wid & 1;
  int rowA  = mt * 16 + (lane & 15);
  int base8 = (lane >> 4) * 8;

  v8f acc = {};
  for (int kc = 0; kc < 16; ++kc) {
    const _Float16* ap = H_all + (size_t)rowA * HIDn + kc * 32 + base8;
    v16h a = load_frag_a(ap, ap + 16);
    v16h b = *(const v16h*)(fcp + ((size_t)(nt * 16 + kc) * 32 + lane) * 16);
    acc = __builtin_amdgcn_wmma_f32_16x16x32_f16(false, a, false, b, (short)0, acc, false, false);
  }
  int ncol = nt * 16 + (lane & 15);
  float bb = fcb[ncol];
#pragma unroll
  for (int r = 0; r < 8; ++r) {
    int grow = mt * 16 + r + ((lane >> 4) << 3);    // = b*T + t
    em[(size_t)grow * Kt + ncol] = acc[r] + bb;
  }
}

// Gold path score per batch element
__global__ void gold_score(const int* __restrict__ y,
                           const float* __restrict__ em,
                           const float* __restrict__ trans,
                           const float* __restrict__ start,
                           const float* __restrict__ endv,
                           float* __restrict__ num) {
  int b = blockIdx.x;
  int tid = threadIdx.x;
  __shared__ float red[256];
  float p = 0.0f;
  for (int t = tid; t < Tn; t += 256) {
    int yt = y[(size_t)b * Tn + t];
    p += em[((size_t)b * Tn + t) * Kt + yt];
    if (t + 1 < Tn) p += trans[(size_t)yt * Kt + y[(size_t)b * Tn + t + 1]];
  }
  red[tid] = p;
  __syncthreads();
  for (int s = 128; s > 0; s >>= 1) {
    if (tid < s) red[tid] += red[tid + s];
    __syncthreads();
  }
  if (tid == 0)
    num[b] = red[0] + start[y[(size_t)b * Tn]] + endv[y[(size_t)b * Tn + Tn - 1]];
}

// CRF forward algorithm (log-partition) + final loss. One workgroup, 1024 threads.
__global__ __launch_bounds__(1024)
void crf_forward(const float* __restrict__ em,
                 const float* __restrict__ trans,
                 const float* __restrict__ start,
                 const float* __restrict__ endv,
                 const float* __restrict__ num,
                 float* __restrict__ out) {
  __shared__ float tr[Kt * Kt];
  __shared__ float A0[Bn * Kt];
  __shared__ float A1[Bn * Kt];
  __shared__ float lz[Bn];
  int tid = threadIdx.x;
  tr[tid] = trans[tid];                      // 1024 == 32*32 exactly
  for (int s = tid; s < Bn * Kt; s += 1024) {
    int b = s >> 5, k = s & 31;
    A0[s] = start[k] + em[((size_t)b * Tn) * Kt + k];
  }
  __syncthreads();
  float* cur = A0;
  float* nxt = A1;
  for (int t = 1; t < Tn; ++t) {
    for (int s = tid; s < Bn * Kt; s += 1024) {
      int b = s >> 5, kp = s & 31;
      float m = -1e30f;
#pragma unroll
      for (int k = 0; k < Kt; ++k) m = fmaxf(m, cur[b * Kt + k] + tr[k * Kt + kp]);
      float sum = 0.0f;
#pragma unroll
      for (int k = 0; k < Kt; ++k) sum += __expf(cur[b * Kt + k] + tr[k * Kt + kp] - m);
      nxt[s] = m + __logf(sum) + em[((size_t)b * Tn + t) * Kt + kp];
    }
    __syncthreads();
    float* tmp = cur; cur = nxt; nxt = tmp;
  }
  if (tid < Bn) {
    int b = tid;
    float m = -1e30f;
#pragma unroll
    for (int k = 0; k < Kt; ++k) m = fmaxf(m, cur[b * Kt + k] + endv[k]);
    float sum = 0.0f;
#pragma unroll
    for (int k = 0; k < Kt; ++k) sum += __expf(cur[b * Kt + k] + endv[k] - m);
    lz[b] = m + __logf(sum);
  }
  __syncthreads();
  if (tid == 0) {
    float acc = 0.0f;
    for (int b = 0; b < Bn; ++b) acc += lz[b] - num[b];
    out[0] = acc / (float)Bn;
  }
}

// ---------------------------------------------------------------------------
extern "C" void kernel_launch(void* const* d_in, const int* in_sizes, int n_in,
                              void* d_out, int out_size, void* d_ws, size_t ws_size,
                              hipStream_t stream) {
  (void)in_sizes; (void)n_in; (void)out_size; (void)ws_size;
  const int*   X     = (const int*)d_in[0];
  const int*   y     = (const int*)d_in[1];
  const float* embed = (const float*)d_in[2];
  const float* Wi_f  = (const float*)d_in[3];
  const float* Wh_f  = (const float*)d_in[4];
  const float* b_f   = (const float*)d_in[5];
  const float* Wi_b  = (const float*)d_in[6];
  const float* Wh_b  = (const float*)d_in[7];
  const float* b_b   = (const float*)d_in[8];
  const float* fcW   = (const float*)d_in[9];
  const float* fcb   = (const float*)d_in[10];
  const float* trans = (const float*)d_in[11];
  const float* start = (const float*)d_in[12];
  const float* endv  = (const float*)d_in[13];

  char* ws = (char*)d_ws;
  size_t off = 0;
  auto alloc = [&](size_t bytes) {
    size_t o = off;
    off += (bytes + 255) & ~(size_t)255;
    return o;
  };
  _Float16* xf     = (_Float16*)(ws + alloc((size_t)Bn * Tn * En * 2));       // 16 MB
  _Float16* wpack  = (_Float16*)(ws + alloc((size_t)2 * 1024 * 512 * 2));     // 2 MB
  _Float16* fcp    = (_Float16*)(ws + alloc((size_t)32 * 512 * 2));           // 32 KB
  _Float16* Hall   = (_Float16*)(ws + alloc((size_t)Bn * Tn * HIDn * 2));     // 32 MB
  _Float16* hstate = (_Float16*)(ws + alloc((size_t)2 * 2 * Bn * Hn * 2));    // [parity][dir][B][H]
  float*    cstate = (float*)   (ws + alloc((size_t)2 * Bn * Hn * 4));        // [dir][B][H]
  float*    em     = (float*)   (ws + alloc((size_t)Bn * Tn * Kt * 4));       // 4 MB
  float*    num    = (float*)   (ws + alloc((size_t)Bn * 4));

  hipMemsetAsync(hstate, 0, (size_t)2 * 2 * Bn * Hn * 2, stream);
  hipMemsetAsync(cstate, 0, (size_t)2 * Bn * Hn * 4, stream);

  pack_lstm_weights<<<128, 256, 0, stream>>>(Wi_f, Wh_f, wpack);
  pack_lstm_weights<<<128, 256, 0, stream>>>(Wi_b, Wh_b, wpack + (size_t)1024 * 512);
  pack_fc_weights<<<4, 256, 0, stream>>>(fcW, fcp);
  embed_gather<<<(Bn * Tn * En) / 256, 256, 0, stream>>>(X, embed, xf);

  const size_t hbuf = (size_t)2 * Bn * Hn;  // one parity's [dir][B][H]
  for (int t = 0; t < Tn; ++t) {
    const _Float16* h_in  = hstate + (size_t)(t & 1) * hbuf;
    _Float16*       h_out = hstate + (size_t)((t + 1) & 1) * hbuf;
    lstm_step<<<dim3(16, 2), 128, 0, stream>>>(t, xf, wpack, b_f, b_b,
                                               h_in, h_out, cstate, Hall);
  }

  fc_emissions<<<512, 256, 0, stream>>>(Hall, fcp, fcb, em);
  gold_score<<<Bn, 256, 0, stream>>>(y, em, trans, start, endv, num);
  crf_forward<<<1, 1024, 0, stream>>>(em, trans, start, endv, num, (float*)d_out);
}
